// TorchLMHeadMSE_28681791602728
// MI455X (gfx1250) — compile-verified, matrix operand order checked by reading, not run on hardware
//
#include <hip/hip_runtime.h>

// ---------------------------------------------------------------------------
// LM-head + MSE:  mean((x @ W^T - y)^2)
//   x: [2048, 2048] f32   y: [2048, 32000] f32   W: [32000, 2048] f32
// f16 WMMA GEMM (f32 accum), fused squared-error epilogue, deterministic
// two-stage reduction via d_ws partials.
// ---------------------------------------------------------------------------

typedef __attribute__((ext_vector_type(16))) _Float16 v16h;
typedef __attribute__((ext_vector_type(8)))  float    v8f;

#define GN 2048     // rows of x / logits
#define GH 2048     // K dimension (H)
#define GV 32000    // rows of W / cols of logits

#define BM 128      // block tile over N
#define BN 128      // block tile over V
#define BK 64       // K chunk staged in LDS
#define LDK 72      // padded LDS row stride in halfs (144 B, 16B-aligned rows)

// Load one 16x32 f16 WMMA operand fragment from LDS (ISA 7.12.2 layout):
// lanes 0-15 hold row (base+lr), K = [ks,ks+8) and [ks+16,ks+24)
// lanes 16-31 hold row (base+lr), K = [ks+8,ks+16) and [ks+24,ks+32)
__device__ __forceinline__ v16h load_frag_lds(const _Float16* buf, int rowbase,
                                              int ks, int half, int lr) {
  const _Float16* p = buf + (rowbase + lr) * LDK + ks + half * 8;
  union { v16h v; uint4 q[2]; } u;
  u.q[0] = *(const uint4*)(p);
  u.q[1] = *(const uint4*)(p + 16);
  return u.v;
}

// Convert 4 f32 -> 4 f16 and store packed (ds_store_b64).
__device__ __forceinline__ void store_h4(_Float16* dst, float4 v) {
  union { _Float16 h[4]; unsigned long long u; } p;
  p.h[0] = (_Float16)v.x; p.h[1] = (_Float16)v.y;
  p.h[2] = (_Float16)v.z; p.h[3] = (_Float16)v.w;
  *(unsigned long long*)dst = p.u;
}

__global__ __launch_bounds__(256, 2)
void lmhead_mse_gemm(const float* __restrict__ x, const float* __restrict__ y,
                     const float* __restrict__ W, float* __restrict__ partials) {
  __shared__ __align__(16) _Float16 As[BM * LDK];   // 18 KB
  __shared__ __align__(16) _Float16 Bs[BN * LDK];   // 18 KB
  __shared__ float red[8];

  const int t    = threadIdx.x;
  const int lane = t & 31;          // wave32
  const int w    = t >> 5;          // 8 waves / block
  const int wrow = w >> 1;          // 0..3 -> 32-row strip
  const int wcol = w & 1;           // 0..1 -> 64-col strip
  const int half = lane >> 4;
  const int lr   = lane & 15;

  const int m0 = blockIdx.x * BM;   // over N (x rows)
  const int n0 = blockIdx.y * BN;   // over V (W rows)

  v8f acc[2][4] = {};               // 32x64 logits per wave

  const float* xrow = x + (long)m0 * GH;
  const float* wrw  = W + (long)n0 * GH;

  for (int kc = 0; kc < GH; kc += BK) {
    __syncthreads();                // previous chunk fully consumed
    // Stage 128x64 f32 of each operand -> f16 in LDS.
    // 2048 float4 per operand, 8 per thread.
#pragma unroll
    for (int it = 0; it < 8; ++it) {
      const int c   = t + 256 * it;
      const int row = c >> 4;
      const int col = (c & 15) << 2;
      const long go = (long)row * GH + kc + col;
      float4 va = *(const float4*)(xrow + go);
      float4 vb = *(const float4*)(wrw  + go);
      if (kc + BK < GH) {           // pull next chunk toward L2/L0
        __builtin_prefetch(xrow + go + BK, 0, 0);
        __builtin_prefetch(wrw  + go + BK, 0, 0);
      }
      store_h4(&As[row * LDK + col], va);
      store_h4(&Bs[row * LDK + col], vb);
    }
    __syncthreads();                // chunk visible to all waves

#pragma unroll
    for (int ks = 0; ks < BK; ks += 32) {
      v16h a0 = load_frag_lds(As, 32 * wrow,      ks, half, lr);
      v16h a1 = load_frag_lds(As, 32 * wrow + 16, ks, half, lr);
      v16h b0 = load_frag_lds(Bs, 64 * wcol,      ks, half, lr);
      v16h b1 = load_frag_lds(Bs, 64 * wcol + 16, ks, half, lr);
      v16h b2 = load_frag_lds(Bs, 64 * wcol + 32, ks, half, lr);
      v16h b3 = load_frag_lds(Bs, 64 * wcol + 48, ks, half, lr);
      acc[0][0] = __builtin_amdgcn_wmma_f32_16x16x32_f16(false, a0, false, b0, (short)0, acc[0][0], false, false);
      acc[0][1] = __builtin_amdgcn_wmma_f32_16x16x32_f16(false, a0, false, b1, (short)0, acc[0][1], false, false);
      acc[0][2] = __builtin_amdgcn_wmma_f32_16x16x32_f16(false, a0, false, b2, (short)0, acc[0][2], false, false);
      acc[0][3] = __builtin_amdgcn_wmma_f32_16x16x32_f16(false, a0, false, b3, (short)0, acc[0][3], false, false);
      acc[1][0] = __builtin_amdgcn_wmma_f32_16x16x32_f16(false, a1, false, b0, (short)0, acc[1][0], false, false);
      acc[1][1] = __builtin_amdgcn_wmma_f32_16x16x32_f16(false, a1, false, b1, (short)0, acc[1][1], false, false);
      acc[1][2] = __builtin_amdgcn_wmma_f32_16x16x32_f16(false, a1, false, b2, (short)0, acc[1][2], false, false);
      acc[1][3] = __builtin_amdgcn_wmma_f32_16x16x32_f16(false, a1, false, b3, (short)0, acc[1][3], false, false);
    }
  }

  // Fused epilogue: (logit - y)^2, accumulate per-thread.
  // C/D layout: VGPR r, lanes 0-15 -> (M = r,    N = lane)
  //                     lanes 16-31 -> (M = r+8, N = lane-16)
  float sum = 0.0f;
#pragma unroll
  for (int i = 0; i < 2; ++i) {
#pragma unroll
    for (int j = 0; j < 4; ++j) {
      const int gm = m0 + 32 * wrow + 16 * i + 8 * half;
      const int gn = n0 + 64 * wcol + 16 * j + lr;
      const float* yp = y + (long)gm * GV + gn;
#pragma unroll
      for (int r = 0; r < 8; ++r) {
        float d = acc[i][j][r] - yp[(long)r * GV];
        sum = fmaf(d, d, sum);
      }
    }
  }

  // Wave reduction (wave32), then LDS across the 8 waves.
#pragma unroll
  for (int off = 16; off > 0; off >>= 1) sum += __shfl_xor(sum, off, 32);
  if (lane == 0) red[w] = sum;
  __syncthreads();
  if (t == 0) {
    float s = 0.0f;
#pragma unroll
    for (int k = 0; k < 8; ++k) s += red[k];
    partials[blockIdx.y * gridDim.x + blockIdx.x] = s;
  }
}

__global__ __launch_bounds__(256)
void reduce_mean(const float* __restrict__ partials, int n,
                 float* __restrict__ out, float scale) {
  __shared__ float red[8];
  float s = 0.0f;
  for (int i = threadIdx.x; i < n; i += 256) s += partials[i];
#pragma unroll
  for (int off = 16; off > 0; off >>= 1) s += __shfl_xor(s, off, 32);
  const int lane = threadIdx.x & 31, w = threadIdx.x >> 5;
  if (lane == 0) red[w] = s;
  __syncthreads();
  if (threadIdx.x == 0) {
    float tot = 0.0f;
#pragma unroll
    for (int k = 0; k < 8; ++k) tot += red[k];
    out[0] = tot * scale;
  }
}

extern "C" void kernel_launch(void* const* d_in, const int* in_sizes, int n_in,
                              void* d_out, int out_size, void* d_ws, size_t ws_size,
                              hipStream_t stream) {
  (void)in_sizes; (void)n_in; (void)out_size; (void)ws_size;
  const float* x = (const float*)d_in[0];   // [GN, GH]
  const float* y = (const float*)d_in[1];   // [GN, GV]
  const float* W = (const float*)d_in[2];   // [GV, GH]
  float* partials = (float*)d_ws;           // (GN/BM)*(GV/BN) = 4000 floats

  dim3 grid(GN / BM, GV / BN);              // 16 x 250; x-major -> W slab reuse in L2
  lmhead_mse_gemm<<<grid, dim3(256), 0, stream>>>(x, y, W, partials);

  const int nPart = (GN / BM) * (GV / BN);
  const float scale = (float)(1.0 / ((double)GN * (double)GV));
  reduce_mean<<<dim3(1), dim3(256), 0, stream>>>(partials, nPart, (float*)d_out, scale);
}
